// StructureExtractor_46669114638806
// MI455X (gfx1250) — compile-verified, matrix operand order checked by reading, not run on hardware
//
#include <hip/hip_runtime.h>
#include <hip/hip_bf16.h>

// ---------------------------------------------------------------------------
// Problem constants (from reference)
// ---------------------------------------------------------------------------
#define NB     16          // batch
#define LTOK   4800        // H_C8*W_C8
#define L3D    38400
#define NA     64          // anchors
#define DC     256         // D_COLOR
#define DIN    512         // D_IN = DC + 4*A
#define W_C8   80
#define SCALE  8

// WMMA fragment helpers (gfx1250, wave32)
typedef _Float16 v16h __attribute__((ext_vector_type(16)));
typedef _Float16 v8h  __attribute__((ext_vector_type(8)));
typedef float    v8f  __attribute__((ext_vector_type(8)));

union Frag16 { v16h v; v8h h[2]; };

#define XPITCH 520   // 512 + 8 halfs padding -> 1040B row stride, bank-conflict free

// ---------------------------------------------------------------------------
// Pack W1 (512x512 f32, row = input dim) into WMMA B-fragment order, f16.
// Layout: [kt (0..15)][nt (0..31)][lane (0..31)][elem (0..15)]
// elem i<8  -> k = kt*32 + base8 + i        (base8 = (lane>=16)*8)
// elem i>=8 -> k = kt*32 + 16 + base8 + i-8
// column    = nt*16 + (lane&15)
// ---------------------------------------------------------------------------
__global__ void pack_w1_kernel(const float* __restrict__ W1, _Float16* __restrict__ W1p) {
  int t = blockIdx.x * 256 + threadIdx.x;            // 16*32*32*16 = 262144
  int i    = t & 15;
  int lane = (t >> 4) & 31;
  int nt   = (t >> 9) & 31;
  int kt   = t >> 14;
  int base8 = (lane >> 4) << 3;
  int klocal = (i < 8) ? (base8 + i) : (16 + base8 + (i - 8));
  int k   = kt * 32 + klocal;
  int col = (nt << 4) + (lane & 15);
  W1p[t] = (_Float16)W1[k * DIN + col];
}

__global__ void pack_w2_kernel(const float* __restrict__ W2, _Float16* __restrict__ W2p) {
  int t = blockIdx.x * 256 + threadIdx.x;            // 16*16*32*16 = 131072
  int i    = t & 15;
  int lane = (t >> 4) & 31;
  int nt   = (t >> 9) & 15;
  int kt   = t >> 13;
  int base8 = (lane >> 4) << 3;
  int klocal = (i < 8) ? (base8 + i) : (16 + base8 + (i - 8));
  int k   = kt * 32 + klocal;
  int col = (nt << 4) + (lane & 15);
  W2p[t] = (_Float16)W2[k * DC + col];
}

// ---------------------------------------------------------------------------
// Structured info: per token, 64 anchors -> [dx,dy,dz,dist] normalized by
// sum(|.|) over anchors per channel; channel-major output (4 x 64) as f16.
// grid: (ceil(L/256), N, 2 streams), block 256
// ---------------------------------------------------------------------------
__global__ void struct_kernel(const float* __restrict__ pts0,
                              const float* __restrict__ pts1,
                              const int*   __restrict__ anc_i,
                              const int*   __restrict__ anc_j,
                              _Float16*    __restrict__ S) {
  const int n      = blockIdx.y;
  const int stream = blockIdx.z;
  const float* pts = stream ? pts1 : pts0;
  const int*   ids = stream ? anc_j : anc_i;

  __shared__ float ax[NA], ay[NA], az[NA];
  const int tid = threadIdx.x;
  if (tid < NA) {
    int id = ids[n * NA + tid];
    const float* p = pts + ((size_t)n * L3D + (size_t)id) * 3;
    ax[tid] = p[0]; ay[tid] = p[1]; az[tid] = p[2];
  }
  __syncthreads();

  int l = blockIdx.x * 256 + tid;
  if (l >= LTOK) return;

  int ci = 640 * (l / W_C8) + SCALE * (l % W_C8);     // coarse index into L3D
  const float* p = pts + ((size_t)n * L3D + (size_t)ci) * 3;
  float px = p[0], py = p[1], pz = p[2];

  float sx = 0.f, sy = 0.f, sz = 0.f, sd = 0.f;
  #pragma unroll 4
  for (int a = 0; a < NA; ++a) {
    float dx = px - ax[a], dy = py - ay[a], dz = pz - az[a];
    float dist = dx * dx + dy * dy + dz * dz;
    sx += fabsf(dx); sy += fabsf(dy); sz += fabsf(dz); sd += dist;
  }
  float ivx = 1.f / sx, ivy = 1.f / sy, ivz = 1.f / sz, ivd = 1.f / sd;

  _Float16* so = S + ((size_t)stream * NB * LTOK + (size_t)n * LTOK + l) * DC;
  #pragma unroll 4
  for (int a = 0; a < NA; ++a) {
    float dx = px - ax[a], dy = py - ay[a], dz = pz - az[a];
    float dist = dx * dx + dy * dy + dz * dz;
    so[a]            = (_Float16)(dx * ivx);
    so[NA + a]       = (_Float16)(dy * ivy);
    so[2 * NA + a]   = (_Float16)(dz * ivz);
    so[3 * NA + a]   = (_Float16)(dist * ivd);
  }
}

// ---------------------------------------------------------------------------
// Fused MLP: per block 64 tokens, 256 threads = 8 waves.
// wave w -> row-group rg = w>>1 (16 rows), column half = w&1.
// GEMM1: h = relu([feat|S] @ W1 + b1)   (512 -> 512)
// GEMM2: out = feat + h @ W2 + b2       (512 -> 256)
// grid: (N*L/64, 2 streams), dynamic LDS = 2 * 64 * XPITCH * 2 bytes
// ---------------------------------------------------------------------------
__global__ __launch_bounds__(256)
void mlp_kernel(const float* __restrict__ feat0,
                const float* __restrict__ feat1,
                const _Float16* __restrict__ S,
                const _Float16* __restrict__ W1p,
                const _Float16* __restrict__ W2p,
                const float* __restrict__ b1,
                const float* __restrict__ b2,
                float* __restrict__ out) {
  extern __shared__ _Float16 smem[];
  _Float16* xs = smem;                 // 64 x XPITCH activations (f16)
  _Float16* hs = smem + 64 * XPITCH;   // 64 x XPITCH hidden (f16)

  const int stream = blockIdx.y;
  const float* feat = stream ? feat1 : feat0;
  const size_t tok_base = (size_t)blockIdx.x * 64;
  const size_t out_off  = (size_t)stream * NB * LTOK * DC;
  const int tid = threadIdx.x;

  // ---- stage x = [feat (f32->f16) | S (f16)] into LDS --------------------
  const _Float16* Sb = S + ((size_t)stream * NB * LTOK + tok_base) * DC;
  for (int q = tid; q < 64 * 64; q += 256) {            // feat as float4
    int row = q >> 6; int c4 = (q & 63) << 2;
    float4 f = *(const float4*)(feat + (tok_base + row) * DC + c4);
    _Float16* d = xs + row * XPITCH + c4;
    d[0] = (_Float16)f.x; d[1] = (_Float16)f.y;
    d[2] = (_Float16)f.z; d[3] = (_Float16)f.w;
  }
  for (int q = tid; q < 64 * 32; q += 256) {            // S as 16B chunks
    int row = q >> 5; int c8 = (q & 31) << 3;
    *(uint4*)(xs + row * XPITCH + DC + c8) = *(const uint4*)(Sb + row * DC + c8);
  }
  __syncthreads();

  const int lane = tid & 31;
  const int w    = tid >> 5;
  const int rg   = w >> 1;            // row group 0..3
  const int half = w & 1;             // column half
  const int mrow = rg * 16 + (lane & 15);
  const int koff = (lane >> 4) << 3;  // 0 or 8
  const int crow = rg * 16 + ((lane >> 4) << 3);  // C-fragment base row

  // ---- preload all 16 A fragments (K = 512) from LDS ---------------------
  Frag16 a[16];
  #pragma unroll
  for (int kt = 0; kt < 16; ++kt) {
    const _Float16* p = xs + mrow * XPITCH + kt * 32 + koff;
    a[kt].h[0] = *(const v8h*)p;
    a[kt].h[1] = *(const v8h*)(p + 16);
  }

  // ---- GEMM1: 16 column tiles per wave -----------------------------------
  for (int ntl = 0; ntl < 16; ++ntl) {
    const int ng = half * 16 + ntl;
    const _Float16* wp = W1p + ((size_t)ng * 32 + lane) * 16;
    v8f acc = {};
    #pragma unroll
    for (int kt = 0; kt < 16; ++kt) {
      Frag16 b;
      const _Float16* q = wp + (size_t)kt * (32 * 32 * 16);
      b.h[0] = *(const v8h*)q;
      b.h[1] = *(const v8h*)(q + 8);
      acc = __builtin_amdgcn_wmma_f32_16x16x32_f16(
          false, a[kt].v, false, b.v, (short)0, acc, false, false);
    }
    const int col = ng * 16 + (lane & 15);
    const float bias = b1[col];
    #pragma unroll
    for (int v = 0; v < 8; ++v) {
      float r = acc[v] + bias;
      r = fmaxf(r, 0.f);
      hs[(crow + v) * XPITCH + col] = (_Float16)r;
    }
  }
  __syncthreads();

  // ---- reload A fragments from hidden tile -------------------------------
  #pragma unroll
  for (int kt = 0; kt < 16; ++kt) {
    const _Float16* p = hs + mrow * XPITCH + kt * 32 + koff;
    a[kt].h[0] = *(const v8h*)p;
    a[kt].h[1] = *(const v8h*)(p + 16);
  }

  // ---- GEMM2: 8 column tiles per wave + residual epilogue ----------------
  for (int ntl = 0; ntl < 8; ++ntl) {
    const int ng = half * 8 + ntl;
    const _Float16* wp = W2p + ((size_t)ng * 32 + lane) * 16;
    v8f acc = {};
    #pragma unroll
    for (int kt = 0; kt < 16; ++kt) {
      Frag16 b;
      const _Float16* q = wp + (size_t)kt * (16 * 32 * 16);
      b.h[0] = *(const v8h*)q;
      b.h[1] = *(const v8h*)(q + 8);
      acc = __builtin_amdgcn_wmma_f32_16x16x32_f16(
          false, a[kt].v, false, b.v, (short)0, acc, false, false);
    }
    const int col = ng * 16 + (lane & 15);
    const float bias = b2[col];
    #pragma unroll
    for (int v = 0; v < 8; ++v) {
      const size_t tok = tok_base + (size_t)(crow + v);
      out[out_off + tok * DC + col] = feat[tok * DC + col] + acc[v] + bias;
    }
  }
}

// ---------------------------------------------------------------------------
// Launch
// ---------------------------------------------------------------------------
extern "C" void kernel_launch(void* const* d_in, const int* in_sizes, int n_in,
                              void* d_out, int out_size, void* d_ws, size_t ws_size,
                              hipStream_t stream) {
  (void)in_sizes; (void)n_in; (void)out_size; (void)ws_size;
  const float* feat0 = (const float*)d_in[0];
  const float* feat1 = (const float*)d_in[1];
  const float* pts0  = (const float*)d_in[2];
  const float* pts1  = (const float*)d_in[3];
  const int*   anc_i = (const int*)d_in[4];
  const int*   anc_j = (const int*)d_in[5];
  const float* W1    = (const float*)d_in[6];
  const float* b1    = (const float*)d_in[7];
  const float* W2    = (const float*)d_in[8];
  const float* b2    = (const float*)d_in[9];

  _Float16* ws  = (_Float16*)d_ws;
  _Float16* W1p = ws;                       // 262144 halfs
  _Float16* W2p = ws + 262144;              // 131072 halfs
  _Float16* Sws = ws + 262144 + 131072;     // 2*N*L*256 halfs (~78.6 MB)

  pack_w1_kernel<<<1024, 256, 0, stream>>>(W1, W1p);
  pack_w2_kernel<<<512, 256, 0, stream>>>(W2, W2p);

  dim3 gs((LTOK + 255) / 256, NB, 2);
  struct_kernel<<<gs, 256, 0, stream>>>(pts0, pts1, anc_i, anc_j, Sws);

  dim3 gg((NB * LTOK) / 64, 2);
  size_t shmem = (size_t)2 * 64 * XPITCH * sizeof(_Float16);   // 133,120 B
  mlp_kernel<<<gg, 256, shmem, stream>>>(feat0, feat1, Sws, W1p, W2p, b1, b2,
                                         (float*)d_out);
}
